// MultiHeadAttention_66477503807599
// MI455X (gfx1250) — matrix operand order repaired and checked by reading, not compiled, not run on hardware
//
#include <hip/hip_runtime.h>
#include <hip/hip_bf16.h>
#include <math.h>

// ---------------------------------------------------------------------------
// MHA forward with RoPE for gfx1250 (MI455X): bf16 WMMA, f32 accumulate.
// B=2, S=2048, D=1024, H=16, dk=64.
// ---------------------------------------------------------------------------

typedef __attribute__((ext_vector_type(16))) __bf16       v16bf;
typedef __attribute__((ext_vector_type(8)))  float        v8f;
typedef __attribute__((ext_vector_type(4)))  unsigned int u32x4;
typedef __attribute__((ext_vector_type(4)))  int          v4i;

union FragU { v16bf v; u32x4 q[2]; };
union C16  { u32x4 q; __bf16 h[8]; };

// --- CDNA5 async global->LDS copy (ASYNCcnt) with compile-safe fallback ----
#ifndef USE_ASYNC_LDS
#if defined(__has_builtin)
#if __has_builtin(__builtin_amdgcn_global_load_async_to_lds_b128)
#define USE_ASYNC_LDS 1
#endif
#endif
#endif
#ifndef USE_ASYNC_LDS
#define USE_ASYNC_LDS 0
#endif

__device__ __forceinline__ void copy16(const __bf16* g, __bf16* l) {
#if USE_ASYNC_LDS
    typedef __attribute__((address_space(1))) v4i* gp_t;  // global int4*
    typedef __attribute__((address_space(3))) v4i* lp_t;  // LDS int4*
    __builtin_amdgcn_global_load_async_to_lds_b128(
        (gp_t)(unsigned long long)(uintptr_t)g,
        (lp_t)(unsigned int)(uintptr_t)l, 0, 0);
#else
    *(u32x4*)l = *(const u32x4*)g;
#endif
}

__device__ __forceinline__ void wait_async() {
#if USE_ASYNC_LDS
#if defined(__has_builtin) && __has_builtin(__builtin_amdgcn_s_wait_asynccnt)
    __builtin_amdgcn_s_wait_asynccnt(0);
#else
    asm volatile("s_wait_asynccnt 0x0" ::: "memory");
#endif
#endif
}

__device__ inline v8f v8f_zero() {
    v8f z = {0.f, 0.f, 0.f, 0.f, 0.f, 0.f, 0.f, 0.f};
    return z;
}

// B-fragment (32x16 bf16, K x N): lane holds column n = lane&15,
// K range = (lane>>4)*16 .. +15 -> 16 contiguous bf16 (32 B) from the
// row-major "N x K" operand. 16B-aligned loads.
__device__ inline v16bf ld_b_frag(const __bf16* p) {
    FragU f;
    const u32x4* q = (const u32x4*)p;
    f.q[0] = q[0];
    f.q[1] = q[1];
    return f.v;
}

// A-fragment (16x32 bf16, M x K): row m = lane&15, element s holds k = (lane>>4)*16 + s (contiguous order).
__device__ inline v16bf ld_a_frag(const __bf16* base, int ld, int lane) {
    const int m = lane & 15;
    const int half = lane >> 4;
    const __bf16* rp = base + m * ld;
    v16bf a;
#pragma unroll
    for (int v = 0; v < 8; ++v) {
        const int kb = half * 16 + (v << 1);
        a[2 * v]     = rp[kb];
        a[2 * v + 1] = rp[kb + 1];
    }
    return a;
}

__device__ inline v8f wmma_bf16(v16bf a, v16bf b, v8f c) {
    return __builtin_amdgcn_wmma_f32_16x16x32_bf16(
        false, a, false, b, (short)0, c, false, false);
}

// ---------------------------------------------------------------------------
// f32 -> bf16 conversion
// ---------------------------------------------------------------------------
__global__ __launch_bounds__(256) void k_f32_to_bf16(const float* __restrict__ in,
                                                     __bf16* __restrict__ out, int n) {
    int i = (blockIdx.x * 256 + threadIdx.x) * 4;
    if (i + 3 < n) {
        float4 v = *(const float4*)(in + i);
        out[i]     = (__bf16)v.x;
        out[i + 1] = (__bf16)v.y;
        out[i + 2] = (__bf16)v.z;
        out[i + 3] = (__bf16)v.w;
    } else {
        for (int j = 0; j < 4; ++j)
            if (i + j < n) out[i + j] = (__bf16)in[i + j];
    }
}

// ---------------------------------------------------------------------------
// Tiled bf16 GEMM:  out[M,N] = A[M,K] * Bt[N,K]^T   (i.e. x @ W^T)
// Block tile 128x128, K-step 32, 256 threads = 8 waves (4 M x 2 N);
// wave tile 32x64 = 2x4 WMMA tiles. Double-buffered LDS with async
// global->LDS copies: one barrier per K-step, copy overlaps WMMA.
// ---------------------------------------------------------------------------
template <bool STORE_BF16>
__global__ __launch_bounds__(256) void k_gemm(const __bf16* __restrict__ A,
                                              const __bf16* __restrict__ Bt,
                                              void* __restrict__ outv,
                                              int M, int N, int K) {
    __shared__ __bf16 sA[2][128 * 40];  // 32 halfs/row padded to 40
    __shared__ __bf16 sB[2][128 * 40];

    const int tid  = threadIdx.x;
    const int lane = tid & 31;
    const int wid  = tid >> 5;
    const int wm   = wid & 3;   // 4 waves in M, 32 rows each
    const int wn   = wid >> 2;  // 2 waves in N, 64 cols each
    const int m0   = blockIdx.x * 128;
    const int n0   = blockIdx.y * 128;
    const int n15  = lane & 15;
    const int h16  = (lane >> 4) << 4;
    const int rhi  = (lane >> 4) * 8;

    v8f acc[2][4];
#pragma unroll
    for (int i = 0; i < 2; ++i)
#pragma unroll
        for (int j = 0; j < 4; ++j) acc[i][j] = v8f_zero();

    // A tile 128x32 = 512 x 16B chunks (2/thread); B likewise.
    auto stage = [&](int buf, int kc) {
#pragma unroll
        for (int i = 0; i < 2; ++i) {
            const int c = tid + i * 256, r = c >> 2, sg = (c & 3) << 3;
            copy16(A + (size_t)(m0 + r) * K + kc + sg, &sA[buf][r * 40 + sg]);
            copy16(Bt + (size_t)(n0 + r) * K + kc + sg, &sB[buf][r * 40 + sg]);
        }
    };

    stage(0, 0);
    int ibuf = 0;
    for (int kc = 0; kc < K; kc += 32, ibuf ^= 1) {
        wait_async();        // this wave's copies into sX[ibuf] done
        __syncthreads();     // all waves' copies done; prev reads done
        if (kc + 32 < K) stage(ibuf ^ 1, kc + 32);  // overlaps compute

        const __bf16* cA = sA[ibuf];
        const __bf16* cB = sB[ibuf];
        v16bf af[2], bfr[4];
#pragma unroll
        for (int t = 0; t < 2; ++t)
            af[t] = ld_a_frag(cA + (wm * 32 + t * 16) * 40, 40, lane);
#pragma unroll
        for (int nt = 0; nt < 4; ++nt)
            bfr[nt] = ld_b_frag(cB + (wn * 64 + nt * 16 + n15) * 40 + h16);
#pragma unroll
        for (int nt = 0; nt < 4; ++nt)
#pragma unroll
            for (int mt = 0; mt < 2; ++mt)
                acc[mt][nt] = wmma_bf16(af[mt], bfr[nt], acc[mt][nt]);
    }

#pragma unroll
    for (int mt = 0; mt < 2; ++mt)
#pragma unroll
        for (int nt = 0; nt < 4; ++nt)
#pragma unroll
            for (int r = 0; r < 8; ++r) {
                const int row = m0 + wm * 32 + mt * 16 + rhi + r;
                const int col = n0 + wn * 64 + nt * 16 + n15;
                const float v = acc[mt][nt][r];
                if (STORE_BF16)
                    ((__bf16*)outv)[(size_t)row * N + col] = (__bf16)v;
                else
                    ((float*)outv)[(size_t)row * N + col] = v;
            }
}

// ---------------------------------------------------------------------------
// RoPE + head-split permute: (B*S,1024) bf16 -> (B*H,S,64) bf16 for Q,K;
// plain permute for V. One thread per (b,h,s,i<32) rotation pair.
// ---------------------------------------------------------------------------
__global__ __launch_bounds__(256) void k_rope_permute(const __bf16* __restrict__ Pq,
                                                      const __bf16* __restrict__ Pk,
                                                      const __bf16* __restrict__ Pv,
                                                      __bf16* __restrict__ Qh,
                                                      __bf16* __restrict__ Kh,
                                                      __bf16* __restrict__ Vh) {
    const int t  = blockIdx.x * 256 + threadIdx.x;
    const int i  = t & 31;
    const int s  = (t >> 5) & 2047;
    const int bh = t >> 16;
    const int b  = bh >> 4;
    const int h  = bh & 15;

    const size_t prow = ((size_t)b * 2048 + s) * 1024 + h * 64;
    const size_t orow = ((size_t)bh * 2048 + s) * 64;

    const float invf = __expf(-(float)i * (9.210340371976184f / 32.f));
    const float fr   = (float)s * invf;
    float sn, cs;
    sincosf(fr, &sn, &cs);

    {
        const float x1 = (float)Pq[prow + i];
        const float x2 = (float)Pq[prow + 32 + i];
        Qh[orow + i]      = (__bf16)(x1 * cs - x2 * sn);
        Qh[orow + 32 + i] = (__bf16)(x2 * cs + x1 * sn);
    }
    {
        const float x1 = (float)Pk[prow + i];
        const float x2 = (float)Pk[prow + 32 + i];
        Kh[orow + i]      = (__bf16)(x1 * cs - x2 * sn);
        Kh[orow + 32 + i] = (__bf16)(x2 * cs + x1 * sn);
    }
    Vh[orow + i]      = Pv[prow + i];
    Vh[orow + 32 + i] = Pv[prow + 32 + i];
}

// ---------------------------------------------------------------------------
// One key-block step of flash attention. MASK instantiation only for the
// diagonal block. Fragments are hoisted so WMMAs issue back-to-back after
// a single LDS-load clause instead of load->wait->wmma chains.
// ---------------------------------------------------------------------------
template <bool MASK>
__device__ __forceinline__ void flash_step(int kk0, int q0, int lane, int n15,
                                           int rhi, int h16,
                                           const __bf16* sK, const __bf16* sVt,
                                           __bf16* pw, const v16bf* aQ,
                                           v8f* accV, float* mrow, float* lrow) {
    v8f sc[4];
#pragma unroll
    for (int i = 0; i < 4; ++i) sc[i] = v8f_zero();
#pragma unroll
    for (int ks = 0; ks < 2; ++ks) {
        v16bf bk[4];
#pragma unroll
        for (int nt = 0; nt < 4; ++nt)
            bk[nt] = ld_b_frag(sK + (nt * 16 + n15) * 72 + ks * 32 + h16);
#pragma unroll
        for (int nt = 0; nt < 4; ++nt)
            sc[nt] = wmma_bf16(aQ[ks], bk[nt], sc[nt]);
    }

    float p[4][8], alpha[8];
#pragma unroll
    for (int r = 0; r < 8; ++r) {
        const int qrow = q0 + rhi + r;
        float rm = -1e30f;
#pragma unroll
        for (int nt = 0; nt < 4; ++nt) {
            float x = sc[nt][r] * 0.125f;  // 1/sqrt(64)
            if (MASK && (kk0 + nt * 16 + n15 > qrow)) x = -1e30f;
            p[nt][r] = x;
            rm = fmaxf(rm, x);
        }
#pragma unroll
        for (int off = 8; off > 0; off >>= 1)
            rm = fmaxf(rm, __shfl_xor(rm, off, 32));
        const float nm = fmaxf(mrow[r], rm);
        alpha[r] = __expf(mrow[r] - nm);
        mrow[r]  = nm;
        float rs = 0.f;
#pragma unroll
        for (int nt = 0; nt < 4; ++nt) {
            const float e = __expf(p[nt][r] - nm);
            p[nt][r] = e;
            rs += e;
        }
#pragma unroll
        for (int off = 8; off > 0; off >>= 1)
            rs += __shfl_xor(rs, off, 32);
        lrow[r] = lrow[r] * alpha[r] + rs;
    }
#pragma unroll
    for (int nt = 0; nt < 4; ++nt)
#pragma unroll
        for (int r = 0; r < 8; ++r) accV[nt][r] *= alpha[r];

    // Re-layout P (D-fragment) -> A-fragment via wave-local LDS round trip
#pragma unroll
    for (int nt = 0; nt < 4; ++nt)
#pragma unroll
        for (int r = 0; r < 8; ++r)
            pw[(rhi + r) * 72 + nt * 16 + n15] = (__bf16)p[nt][r];

#pragma unroll
    for (int ks = 0; ks < 2; ++ks) {
        const v16bf aP = ld_a_frag(pw + ks * 32, 72, lane);
        v16bf bv[4];
#pragma unroll
        for (int nt = 0; nt < 4; ++nt)
            bv[nt] = ld_b_frag(sVt + (nt * 16 + n15) * 72 + ks * 32 + h16);
#pragma unroll
        for (int nt = 0; nt < 4; ++nt)
            accV[nt] = wmma_bf16(aP, bv[nt], accV[nt]);
    }
}

// ---------------------------------------------------------------------------
// Flash attention: workgroup = (b,h) x 64 query rows; 4 waves x 16 rows.
// K blocks double-buffered into LDS with async copies; V blocks register-
// staged and transposed into LDS. Copy of block kb+1 overlaps compute of kb.
// ---------------------------------------------------------------------------
__global__ __launch_bounds__(128) void k_flash(const __bf16* __restrict__ Qh,
                                               const __bf16* __restrict__ Kh,
                                               const __bf16* __restrict__ Vh,
                                               __bf16* __restrict__ ctx) {
    __shared__ __bf16 sK[2][64 * 72];  // K block, row-major (keys x dk)
    __shared__ __bf16 sVt[64 * 72];    // V^T block (dk x keys)
    __shared__ __bf16 sP[4 * 16 * 72];

    const int tid  = threadIdx.x;
    const int lane = tid & 31;
    const int wave = tid >> 5;
    const int qb   = blockIdx.x;
    const int bh   = blockIdx.y;
    const int q0   = qb * 64 + wave * 16;

    const size_t base = (size_t)bh * 2048 * 64;
    const __bf16* Qb = Qh + base;
    const __bf16* Kb = Kh + base;
    const __bf16* Vb = Vh + base;

    const int n15 = lane & 15;
    const int rhi = (lane >> 4) * 8;
    const int h16 = (lane >> 4) << 4;

    v16bf aQ[2];
#pragma unroll
    for (int ks = 0; ks < 2; ++ks)
        aQ[ks] = ld_a_frag(Qb + (size_t)q0 * 64 + ks * 32, 64, lane);

    v8f accV[4];
#pragma unroll
    for (int i = 0; i < 4; ++i) accV[i] = v8f_zero();
    float mrow[8], lrow[8];
#pragma unroll
    for (int r = 0; r < 8; ++r) { mrow[r] = -1e30f; lrow[r] = 0.f; }

    __bf16* pw = sP + wave * 16 * 72;

    // 64x64 block = 512 x 16B chunks; 128 threads x 4 chunks each.
    auto stage_k = [&](int buf, int kk0) {
#pragma unroll
        for (int i = 0; i < 4; ++i) {
            const int c = tid + i * 128, row = c >> 3, sg = (c & 7) << 3;
            copy16(Kb + (size_t)(kk0 + row) * 64 + sg, &sK[buf][row * 72 + sg]);
        }
    };
    u32x4 rv[4];
    auto load_v = [&](int kk0) {
#pragma unroll
        for (int i = 0; i < 4; ++i) {
            const int c = tid + i * 128, row = c >> 3, sg = (c & 7) << 3;
            rv[i] = *(const u32x4*)(Vb + (size_t)(kk0 + row) * 64 + sg);
        }
    };

    stage_k(0, 0);
    load_v(0);
    int cur = 0;
    for (int kb = 0; kb <= qb; ++kb) {  // causal: skip fully-masked blocks
        wait_async();
        __syncthreads();  // sK[cur] complete everywhere; prev LDS reads done
        // transpose-store V block into sVt
#pragma unroll
        for (int i = 0; i < 4; ++i) {
            const int c = tid + i * 128, row = c >> 3, sg = (c & 7) << 3;
            C16 cv;
            cv.q = rv[i];
#pragma unroll
            for (int j = 0; j < 8; ++j) sVt[(sg + j) * 72 + row] = cv.h[j];
        }
        __syncthreads();  // sVt visible to all waves
        if (kb < qb) {    // overlap next block's copies with compute
            stage_k(cur ^ 1, (kb + 1) * 64);
            load_v(kb + 1);
            flash_step<false>(kb * 64, q0, lane, n15, rhi, h16, sK[cur], sVt,
                              pw, aQ, accV, mrow, lrow);
        } else {
            flash_step<true>(kb * 64, q0, lane, n15, rhi, h16, sK[cur], sVt,
                             pw, aQ, accV, mrow, lrow);
        }
        cur ^= 1;
    }

    const int b = bh >> 4;
    const int h = bh & 15;
#pragma unroll
    for (int nt = 0; nt < 4; ++nt)
#pragma unroll
        for (int r = 0; r < 8; ++r) {
            const size_t row = (size_t)b * 2048 + q0 + rhi + r;
            ctx[row * 1024 + h * 64 + nt * 16 + n15] =
                (__bf16)(accV[nt][r] / lrow[r]);
        }
}

// ---------------------------------------------------------------------------
// Host launcher. Inputs: query, key, value, mask, W_q, W_k, W_v, W_o.
// Mask input unused: it is exactly the causal tril mask, applied in-kernel.
// ---------------------------------------------------------------------------
extern "C" void kernel_launch(void* const* d_in, const int* in_sizes, int n_in,
                              void* d_out, int out_size, void* d_ws, size_t ws_size,
                              hipStream_t stream) {
    (void)in_sizes; (void)n_in; (void)out_size; (void)ws_size;

    const float* q  = (const float*)d_in[0];
    const float* k  = (const float*)d_in[1];
    const float* v  = (const float*)d_in[2];
    const float* Wq = (const float*)d_in[4];
    const float* Wk = (const float*)d_in[5];
    const float* Wv = (const float*)d_in[6];
    const float* Wo = (const float*)d_in[7];
    float* out = (float*)d_out;

    char* ws = (char*)d_ws;
    const size_t MB = 1024 * 1024;
    __bf16* Wq_bf = (__bf16*)(ws + 0 * MB);   // 2MB each
    __bf16* Wk_bf = (__bf16*)(ws + 2 * MB);
    __bf16* Wv_bf = (__bf16*)(ws + 4 * MB);
    __bf16* Wo_bf = (__bf16*)(ws + 6 * MB);
    __bf16* Abf   = (__bf16*)(ws + 8 * MB);   // 8MB, reused per projection
    __bf16* Pq    = (__bf16*)(ws + 16 * MB);  // 8MB each
    __bf16* Pk    = (__bf16*)(ws + 24 * MB);
    __bf16* Pv    = (__bf16*)(ws + 32 * MB);
    __bf16* Qhd   = (__bf16*)(ws + 40 * MB);
    __bf16* Khd   = (__bf16*)(ws + 48 * MB);
    __bf16* Vhd   = (__bf16*)(ws + 56 * MB);
    __bf16* Ctx   = (__bf16*)(ws + 64 * MB);  // total 72MB

    const int nW = 1024 * 1024;
    const int nX = 4096 * 1024;

    k_f32_to_bf16<<<nW / 1024, 256, 0, stream>>>(Wq, Wq_bf, nW);
    k_f32_to_bf16<<<nW / 1024, 256, 0, stream>>>(Wk, Wk_bf, nW);
    k_f32_to_bf16<<<nW / 1024, 256, 0, stream>>>(Wv, Wv_bf, nW);
    k_f32_to_bf16<<<nW / 1024, 256, 0, stream>>>(Wo, Wo_bf, nW);

    dim3 gg(4096 / 128, 1024 / 128);  // (32,8)

    k_f32_to_bf16<<<nX / 1024, 256, 0, stream>>>(q, Abf, nX);
    k_gemm<true><<<gg, 256, 0, stream>>>(Abf, Wq_bf, Pq, 4096, 1024, 1024);
    k_f32_to_bf16<<<nX / 1024, 256, 0, stream>>>(k, Abf, nX);
    k_gemm<true><<<gg, 256, 0, stream>>>(Abf, Wk_bf, Pk, 4096, 1024, 1024);
    k_f32_to_bf16<<<nX / 1024, 256, 0, stream>>>(v, Abf, nX);
    k_gemm<true><<<gg, 256, 0, stream>>>(Abf, Wv_bf, Pv, 4096, 1024, 1024);

    const int nR = 2 * 16 * 2048 * 32;  // rotation pairs
    k_rope_permute<<<nR / 256, 256, 0, stream>>>(Pq, Pk, Pv, Qhd, Khd, Vhd);

    dim3 fg(2048 / 64, 2 * 16);  // (q-blocks, B*H)
    k_flash<<<fg, 128, 0, stream>>>(Qhd, Khd, Vhd, Ctx);

    k_gemm<false><<<gg, 256, 0, stream>>>(Ctx, Wo_bf, out, 4096, 1024, 1024);
}
